// Block_25744033972482
// MI455X (gfx1250) — matrix operand order repaired and checked by reading, not compiled, not run on hardware
//
#include <hip/hip_runtime.h>
#include <cstdint>
#include <cstddef>

// ---------------- problem constants ----------------
constexpr int BATCH   = 2;
constexpr int SEQ     = 2048;
constexpr int DMODEL  = 512;
constexpr int NHEAD   = 8;
constexpr int DHEAD   = 64;           // DMODEL / NHEAD
constexpr int MROWS   = BATCH * SEQ;  // 4096 rows for every GEMM
constexpr int HID_TOT  = 2730;
constexpr int HID_HALF = 1365;
constexpr int HID_KPAD = 1376;        // 43*32  (K padding for FC2)
constexpr int FC1_NPAD = 2816;        // 22*128 (N padding for FC1)
constexpr float LN_EPS = 1e-5f;

typedef __bf16 bf16;
typedef __attribute__((ext_vector_type(16))) __bf16 v16bf;
typedef __attribute__((ext_vector_type(8)))  float  v8f;
typedef __attribute__((ext_vector_type(4)))  int    v4i;

union Frag { v16bf v; uint4 q[2]; };

__device__ __forceinline__ v8f zero8() {
  v8f z = {0.f, 0.f, 0.f, 0.f, 0.f, 0.f, 0.f, 0.f};
  return z;
}

// D(f32 16x16) = A(bf16 16x32) * B(bf16 32x16) + C
__device__ __forceinline__ v8f wmma_bf16f32(v16bf a, v16bf b, v8f c) {
  return __builtin_amdgcn_wmma_f32_16x16x32_bf16(
      /*neg_a=*/false, a, /*neg_b=*/false, b,
      /*c_mod=*/(short)0, c, /*reuse_a=*/false, /*reuse_b=*/false);
}

// fast math: raw v_sqrt_f32 / v_rcp_f32 (bias term + softmax norm tolerate it)
__device__ __forceinline__ float fast_sqrt(float x) {
#if defined(__gfx1250__) && __has_builtin(__builtin_amdgcn_sqrtf)
  return __builtin_amdgcn_sqrtf(x);
#else
  return sqrtf(x);
#endif
}
__device__ __forceinline__ float fast_rcp(float x) {
#if defined(__gfx1250__) && __has_builtin(__builtin_amdgcn_rcpf)
  return __builtin_amdgcn_rcpf(x);
#else
  return 1.f / x;
#endif
}

// ---- async global->LDS copy (gfx1250), with safe fallback -----------------
#if defined(__gfx1250__) && __has_builtin(__builtin_amdgcn_global_load_async_to_lds_b128)
#define HAVE_ASYNC_LDS 1
#endif

typedef __attribute__((address_space(1))) v4i v4i_g;
typedef __attribute__((address_space(3))) v4i v4i_l;

__device__ __forceinline__ void cp_glds16(void* lds_dst, const void* gsrc) {
#ifdef HAVE_ASYNC_LDS
  __builtin_amdgcn_global_load_async_to_lds_b128(
      (v4i_g*)(__attribute__((address_space(1))) void*)(void*)(uintptr_t)gsrc,
      (v4i_l*)(__attribute__((address_space(3))) void*)lds_dst,
      0, 0);
#else
  *reinterpret_cast<uint4*>(lds_dst) = *reinterpret_cast<const uint4*>(gsrc);
#endif
}

// wait until at most N of this wave's async LDS-loads remain in flight
template <int N>
__device__ __forceinline__ void cp_glds_wait_le() {
#if defined(HAVE_ASYNC_LDS) && __has_builtin(__builtin_amdgcn_s_wait_asynccnt)
  __builtin_amdgcn_s_wait_asynccnt(N);
#endif
}

// ---------------- weight convert + pad to bf16 -----------------------------
__global__ __launch_bounds__(256) void cvt_pad_bf16(
    const float* __restrict__ src, bf16* __restrict__ dst,
    int rs, int cs, int rd, int cd) {
  size_t i = (size_t)blockIdx.x * 256 + threadIdx.x;
  size_t tot = (size_t)rd * cd;
  if (i >= tot) return;
  int r = (int)(i / cd), c = (int)(i % cd);
  float v = (r < rs && c < cs) ? src[(size_t)r * cs + c] : 0.f;
  dst[i] = (bf16)v;
}

// ---------------- LayerNorm (one row of 512 per block) ---------------------
__global__ __launch_bounds__(256) void ln_bf16(
    const float* __restrict__ x, const float* __restrict__ w,
    const float* __restrict__ bb, bf16* __restrict__ out) {
  const int row = blockIdx.x, tid = threadIdx.x;
  const float* xr = x + (size_t)row * DMODEL;
  float v0 = xr[tid], v1 = xr[tid + 256];
  float s = v0 + v1, ss = v0 * v0 + v1 * v1;
#pragma unroll
  for (int o = 16; o > 0; o >>= 1) { s += __shfl_xor(s, o); ss += __shfl_xor(ss, o); }
  __shared__ float rs_[8], rss_[8];
  const int wave = tid >> 5, lane = tid & 31;
  if (lane == 0) { rs_[wave] = s; rss_[wave] = ss; }
  __syncthreads();
  float S = 0.f, SS = 0.f;
#pragma unroll
  for (int i = 0; i < 8; ++i) { S += rs_[i]; SS += rss_[i]; }
  const float mean = S * (1.f / DMODEL);
  const float var  = SS * (1.f / DMODEL) - mean * mean;
  const float inv  = rsqrtf(var + LN_EPS);
  out[(size_t)row * DMODEL + tid]       = (bf16)((v0 - mean) * inv * w[tid] + bb[tid]);
  out[(size_t)row * DMODEL + tid + 256] = (bf16)((v1 - mean) * inv * w[tid + 256] + bb[tid + 256]);
}

// ---------------- tiled bf16 WMMA GEMM:  C = A[M,K] * W[N,K]^T (+epilogue) --
// block tile 128x128, 8 waves -> each wave 32(M) x 64(N), K step 32.
// Double-buffered LDS, pipelined on ASYNCcnt: each thread issues exactly 4
// async b128 loads per tile; async loads complete in order, so waiting
// asynccnt<=4 guarantees the previous tile is resident while the next stays
// in flight.
constexpr int BM = 128, BN = 128, BK = 32;
enum { EPI_BF16 = 0, EPI_F32 = 1, EPI_RES = 2 };

template <int EPI>
__global__ __launch_bounds__(256) void gemm_bf16_wmma(
    const bf16* __restrict__ A,      // [MROWS, K] row-major, K padded to 32
    const bf16* __restrict__ W,      // [Npad, K] row-major (zero padded)
    const float* __restrict__ bias,  // [Nreal]
    const float* __restrict__ resid, // [MROWS, Nreal] (EPI_RES)
    const float* __restrict__ gamma, // [Nreal]        (EPI_RES)
    void* __restrict__ Cout,         // bf16 or f32, ldc = Nreal
    int Nreal, int K) {
  __shared__ __align__(16) bf16 sA[2][BM][BK + 8];
  __shared__ __align__(16) bf16 sB[2][BN][BK + 8];
  const int tid = threadIdx.x;
  const int lane = tid & 31, wave = tid >> 5;
  const int lr = lane & 15, lh = lane >> 4;
  const int wm = wave & 3, wn = wave >> 2;             // 4 (M) x 2 (N) waves
  const int m0 = blockIdx.x * BM, n0 = blockIdx.y * BN;
  const int nTiles = K / BK;

  v8f acc[2][4];
#pragma unroll
  for (int mi = 0; mi < 2; ++mi)
#pragma unroll
    for (int ni = 0; ni < 4; ++ni) acc[mi][ni] = zero8();

  auto issue = [&](int t, int buf) {
    const int k0 = t * BK;
#pragma unroll
    for (int i = 0; i < 2; ++i) {
      const int idx = tid + i * 256;          // 0..511
      const int row = idx >> 2;               // 0..127
      const int col = (idx & 3) * 8;          // 0,8,16,24 (bf16)
      const bf16* ga = A + (size_t)(m0 + row) * K + (k0 + col);
      const bf16* gb = W + (size_t)(n0 + row) * K + (k0 + col);
      cp_glds16(&sA[buf][row][col], ga);
      cp_glds16(&sB[buf][row][col], gb);
      if (k0 + BK < K) {                      // L2 warm-up for tile t+1
        __builtin_prefetch(ga + BK, 0, 1);
        __builtin_prefetch(gb + BK, 0, 1);
      }
    }
  };

  issue(0, 0);
  for (int t = 0; t < nTiles; ++t) {
    const int buf = t & 1;
    if (t + 1 < nTiles) {
      issue(t + 1, buf ^ 1);
      cp_glds_wait_le<4>();   // tile t resident; tile t+1 still in flight
    } else {
      cp_glds_wait_le<0>();
    }
    __syncthreads();

    Frag a[2];
#pragma unroll
    for (int mi = 0; mi < 2; ++mi) {
      const int r = wm * 32 + mi * 16 + lr;
      a[mi].q[0] = *(const uint4*)&sA[buf][r][lh * 8];
      a[mi].q[1] = *(const uint4*)&sA[buf][r][16 + lh * 8];
    }
#pragma unroll
    for (int ni = 0; ni < 4; ++ni) {
      Frag b;
      const int r = wn * 64 + ni * 16 + lr;
      b.q[0] = *(const uint4*)&sB[buf][r][lh * 8];
      b.q[1] = *(const uint4*)&sB[buf][r][16 + lh * 8];
#pragma unroll
      for (int mi = 0; mi < 2; ++mi)
        acc[mi][ni] = wmma_bf16f32(a[mi].v, b.v, acc[mi][ni]);
    }
    __syncthreads();   // all waves done with buf before tile t+2 overwrites it
  }

  // epilogue: C element (lane,reg r): row = r + 8*(lane>>4), col = lane&15
#pragma unroll
  for (int mi = 0; mi < 2; ++mi)
#pragma unroll
    for (int ni = 0; ni < 4; ++ni)
#pragma unroll
      for (int r = 0; r < 8; ++r) {
        const int gm = m0 + wm * 32 + mi * 16 + lh * 8 + r;
        const int gn = n0 + wn * 64 + ni * 16 + lr;
        if (gn < Nreal) {
          float v = acc[mi][ni][r] + bias[gn];
          if constexpr (EPI == EPI_BF16) {
            ((bf16*)Cout)[(size_t)gm * Nreal + gn] = (bf16)v;
          } else if constexpr (EPI == EPI_F32) {
            ((float*)Cout)[(size_t)gm * Nreal + gn] = v;
          } else {
            ((float*)Cout)[(size_t)gm * Nreal + gn] =
                resid[(size_t)gm * Nreal + gn] + gamma[gn] * v;
          }
        }
      }
}

// ---------------- flash attention with distance-ALiBi ----------------------
// grid: (SEQ/128, NHEAD, BATCH); 8 waves, each wave owns 16 queries.
__global__ __launch_bounds__(256) void attn_flash_wmma(
    const bf16* __restrict__ Q, const bf16* __restrict__ Km,
    const bf16* __restrict__ V, const float* __restrict__ coords,
    bf16* __restrict__ ctxOut) {
  __shared__ __align__(16) bf16 sK[32][DHEAD + 8];   // key tile, row-major
  __shared__ __align__(16) bf16 sVT[DHEAD][32 + 8];  // V^T tile
  __shared__ float sCK[32][2];                       // key coords
  __shared__ __align__(16) bf16 sP[8][16][32];       // per-wave P staging

  const int tid = threadIdx.x, lane = tid & 31, wave = tid >> 5;
  const int lr = lane & 15, lh = lane >> 4;
  const int h = blockIdx.y, b = blockIdx.z;
  const int qBase = blockIdx.x * 128 + wave * 16;
  const float slope = exp2f(-(float)(h + 1));  // ALiBi slopes for H=8
  const float scale = 0.125f;                  // 1/sqrt(64)

  // Q fragments (two K=32 chunks of DH=64), loaded straight from global
  Frag aq[2];
  {
    const bf16* qp = Q + (size_t)(b * SEQ + qBase + lr) * DMODEL + h * DHEAD;
#pragma unroll
    for (int c = 0; c < 2; ++c) {
      aq[c].q[0] = *(const uint4*)(qp + c * 32 + lh * 8);
      aq[c].q[1] = *(const uint4*)(qp + c * 32 + 16 + lh * 8);
    }
  }
  // query coords for this lane's 8 rows (row M = r + 8*lh)
  float cqx[8], cqy[8];
#pragma unroll
  for (int r = 0; r < 8; ++r) {
    const int qi = qBase + r + 8 * lh;
    cqx[r] = coords[(size_t)(b * SEQ + qi) * 2 + 0];
    cqy[r] = coords[(size_t)(b * SEQ + qi) * 2 + 1];
  }

  v8f ctx[4];
#pragma unroll
  for (int dn = 0; dn < 4; ++dn) ctx[dn] = zero8();
  float mrow[8], lrow[8];
#pragma unroll
  for (int r = 0; r < 8; ++r) { mrow[r] = -__builtin_inff(); lrow[r] = 0.f; }

  for (int kt = 0; kt < SEQ; kt += 32) {
    {  // cooperative tile loads: 256 threads, 16B each
      const int row = tid >> 3, col = (tid & 7) * 8;
      cp_glds16(&sK[row][col],
                Km + (size_t)(b * SEQ + kt + row) * DMODEL + h * DHEAD + col);
      union { uint4 q; bf16 e[8]; } vv;
      vv.q = *(const uint4*)(V + (size_t)(b * SEQ + kt + row) * DMODEL + h * DHEAD + col);
#pragma unroll
      for (int j = 0; j < 8; ++j) sVT[col + j][row] = vv.e[j];  // transpose
      if (tid < 64)
        sCK[tid >> 1][tid & 1] =
            coords[(size_t)(b * SEQ + kt + (tid >> 1)) * 2 + (tid & 1)];
    }
    cp_glds_wait_le<0>();
    __syncthreads();

    // scores S = scale*(Q K^T) - slope*dist : two 16x16 tiles over 32 keys
    v8f s[2];
#pragma unroll
    for (int nt = 0; nt < 2; ++nt) {
      s[nt] = zero8();
#pragma unroll
      for (int c = 0; c < 2; ++c) {
        Frag bk;
        bk.q[0] = *(const uint4*)&sK[nt * 16 + lr][c * 32 + lh * 8];
        bk.q[1] = *(const uint4*)&sK[nt * 16 + lr][c * 32 + 16 + lh * 8];
        s[nt] = wmma_bf16f32(aq[c].v, bk.v, s[nt]);
      }
#pragma unroll
      for (int r = 0; r < 8; ++r) {
        const float dx = cqx[r] - sCK[nt * 16 + lr][0];
        const float dy = cqy[r] - sCK[nt * 16 + lr][1];
        s[nt][r] = s[nt][r] * scale - slope * fast_sqrt(dx * dx + dy * dy);
      }
    }

    // online softmax: rows live across 16-lane groups (xor<16 stays in group)
    float corr[8];
#pragma unroll
    for (int r = 0; r < 8; ++r) {
      float mv = fmaxf(s[0][r], s[1][r]);
#pragma unroll
      for (int o = 1; o < 16; o <<= 1) mv = fmaxf(mv, __shfl_xor(mv, o));
      const float nm = fmaxf(mrow[r], mv);
      corr[r] = __expf(mrow[r] - nm);
      mrow[r] = nm;
      s[0][r] = __expf(s[0][r] - nm);
      s[1][r] = __expf(s[1][r] - nm);
      float ps = s[0][r] + s[1][r];
#pragma unroll
      for (int o = 1; o < 16; o <<= 1) ps += __shfl_xor(ps, o);
      lrow[r] = lrow[r] * corr[r] + ps;
    }

    // redistribute P (C layout) -> A layout via wave-private LDS
#pragma unroll
    for (int nt = 0; nt < 2; ++nt)
#pragma unroll
      for (int r = 0; r < 8; ++r)
        sP[wave][r + 8 * lh][nt * 16 + lr] = (bf16)s[nt][r];
    Frag ap;
    ap.q[0] = *(const uint4*)&sP[wave][lr][lh * 8];
    ap.q[1] = *(const uint4*)&sP[wave][lr][16 + lh * 8];

    // ctx = ctx*corr + P @ V
#pragma unroll
    for (int dn = 0; dn < 4; ++dn) {
#pragma unroll
      for (int r = 0; r < 8; ++r) ctx[dn][r] *= corr[r];
      Frag bv;
      bv.q[0] = *(const uint4*)&sVT[dn * 16 + lr][lh * 8];
      bv.q[1] = *(const uint4*)&sVT[dn * 16 + lr][16 + lh * 8];
      ctx[dn] = wmma_bf16f32(ap.v, bv.v, ctx[dn]);
    }
    __syncthreads();
  }

#pragma unroll
  for (int dn = 0; dn < 4; ++dn)
#pragma unroll
    for (int r = 0; r < 8; ++r) {
      const int qi = qBase + r + 8 * lh;
      const int dh = dn * 16 + lr;
      ctxOut[(size_t)(b * SEQ + qi) * DMODEL + h * DHEAD + dh] =
          (bf16)(ctx[dn][r] * fast_rcp(lrow[r]));
    }
}

// ---------------- SwiGLU gate: h = silu(h1)*h2, zero-padded to 1376 --------
__global__ __launch_bounds__(256) void swiglu_bf16(
    const float* __restrict__ hpre, bf16* __restrict__ hg) {
  size_t i = (size_t)blockIdx.x * 256 + threadIdx.x;
  if (i >= (size_t)MROWS * HID_KPAD) return;
  const int m = (int)(i / HID_KPAD), j = (int)(i % HID_KPAD);
  float v = 0.f;
  if (j < HID_HALF) {
    const float a = hpre[(size_t)m * HID_TOT + j];
    const float g = hpre[(size_t)m * HID_TOT + HID_HALF + j];
    v = (a * fast_rcp(1.f + __expf(-a))) * g;
  }
  hg[i] = (bf16)v;
}

// ---------------- host orchestration ---------------------------------------
extern "C" void kernel_launch(void* const* d_in, const int* in_sizes, int n_in,
                              void* d_out, int out_size, void* d_ws, size_t ws_size,
                              hipStream_t stream) {
  const float* x      = (const float*)d_in[0];
  const float* coords = (const float*)d_in[1];
  const float* q_w = (const float*)d_in[2];  const float* q_b = (const float*)d_in[3];
  const float* k_w = (const float*)d_in[4];  const float* k_b = (const float*)d_in[5];
  const float* v_w = (const float*)d_in[6];  const float* v_b = (const float*)d_in[7];
  const float* o_w = (const float*)d_in[8];  const float* o_b = (const float*)d_in[9];
  const float* gamma1 = (const float*)d_in[10];
  const float* ln1_w  = (const float*)d_in[11]; const float* ln1_b = (const float*)d_in[12];
  const float* fc1_w  = (const float*)d_in[13]; const float* fc1_b = (const float*)d_in[14];
  const float* fc2_w  = (const float*)d_in[15]; const float* fc2_b = (const float*)d_in[16];
  const float* gamma2 = (const float*)d_in[17];
  const float* ln2_w  = (const float*)d_in[18]; const float* ln2_b = (const float*)d_in[19];
  float* out = (float*)d_out;

  char* ws = (char*)d_ws;
  size_t off = 0;
  auto alloc = [&](size_t bytes) -> char* {
    char* p = ws + off;
    off = (off + bytes + 255) & ~(size_t)255;
    return p;
  };
  bf16* wq   = (bf16*)alloc((size_t)DMODEL * DMODEL * 2);
  bf16* wk   = (bf16*)alloc((size_t)DMODEL * DMODEL * 2);
  bf16* wv   = (bf16*)alloc((size_t)DMODEL * DMODEL * 2);
  bf16* wo   = (bf16*)alloc((size_t)DMODEL * DMODEL * 2);
  bf16* wfc1 = (bf16*)alloc((size_t)FC1_NPAD * DMODEL * 2);
  bf16* wfc2 = (bf16*)alloc((size_t)DMODEL * HID_KPAD * 2);
  bf16* xn   = (bf16*)alloc((size_t)MROWS * DMODEL * 2);
  bf16* qb   = (bf16*)alloc((size_t)MROWS * DMODEL * 2);
  bf16* kb   = (bf16*)alloc((size_t)MROWS * DMODEL * 2);
  bf16* vb   = (bf16*)alloc((size_t)MROWS * DMODEL * 2);
  bf16* ctxb = (bf16*)alloc((size_t)MROWS * DMODEL * 2);
  float* x1  = (float*)alloc((size_t)MROWS * DMODEL * 4);
  bf16* xn2  = (bf16*)alloc((size_t)MROWS * DMODEL * 2);
  float* hpre= (float*)alloc((size_t)MROWS * HID_TOT * 4);
  bf16* hg   = (bf16*)alloc((size_t)MROWS * HID_KPAD * 2);
  (void)ws_size; (void)in_sizes; (void)n_in; (void)out_size;

  const dim3 blk(256);
  auto blocks = [](size_t n) { return (unsigned)((n + 255) / 256); };

  // 1) weight conversion to bf16 (with zero padding where needed)
  cvt_pad_bf16<<<blocks((size_t)DMODEL * DMODEL), blk, 0, stream>>>(q_w, wq, DMODEL, DMODEL, DMODEL, DMODEL);
  cvt_pad_bf16<<<blocks((size_t)DMODEL * DMODEL), blk, 0, stream>>>(k_w, wk, DMODEL, DMODEL, DMODEL, DMODEL);
  cvt_pad_bf16<<<blocks((size_t)DMODEL * DMODEL), blk, 0, stream>>>(v_w, wv, DMODEL, DMODEL, DMODEL, DMODEL);
  cvt_pad_bf16<<<blocks((size_t)DMODEL * DMODEL), blk, 0, stream>>>(o_w, wo, DMODEL, DMODEL, DMODEL, DMODEL);
  cvt_pad_bf16<<<blocks((size_t)FC1_NPAD * DMODEL), blk, 0, stream>>>(fc1_w, wfc1, HID_TOT, DMODEL, FC1_NPAD, DMODEL);
  cvt_pad_bf16<<<blocks((size_t)DMODEL * HID_KPAD), blk, 0, stream>>>(fc2_w, wfc2, DMODEL, HID_HALF, DMODEL, HID_KPAD);

  // 2) LN1
  ln_bf16<<<MROWS, blk, 0, stream>>>(x, ln1_w, ln1_b, xn);

  // 3) Q,K,V projections (bf16 out)
  const dim3 gQKV(MROWS / BM, DMODEL / BN);
  gemm_bf16_wmma<EPI_BF16><<<gQKV, blk, 0, stream>>>(xn, wq, q_b, nullptr, nullptr, qb, DMODEL, DMODEL);
  gemm_bf16_wmma<EPI_BF16><<<gQKV, blk, 0, stream>>>(xn, wk, k_b, nullptr, nullptr, kb, DMODEL, DMODEL);
  gemm_bf16_wmma<EPI_BF16><<<gQKV, blk, 0, stream>>>(xn, wv, v_b, nullptr, nullptr, vb, DMODEL, DMODEL);

  // 4) flash attention with distance ALiBi
  attn_flash_wmma<<<dim3(SEQ / 128, NHEAD, BATCH), blk, 0, stream>>>(qb, kb, vb, coords, ctxb);

  // 5) output projection + residual1:  x1 = x + gamma1*(ctx@o_w^T + o_b)
  gemm_bf16_wmma<EPI_RES><<<gQKV, blk, 0, stream>>>(ctxb, wo, o_b, x, gamma1, x1, DMODEL, DMODEL);

  // 6) LN2
  ln_bf16<<<MROWS, blk, 0, stream>>>(x1, ln2_w, ln2_b, xn2);

  // 7) FC1 (N padded 2816, real 2730)
  gemm_bf16_wmma<EPI_F32><<<dim3(MROWS / BM, FC1_NPAD / BN), blk, 0, stream>>>(
      xn2, wfc1, fc1_b, nullptr, nullptr, hpre, HID_TOT, DMODEL);

  // 8) SwiGLU gate -> bf16, K padded to 1376
  swiglu_bf16<<<blocks((size_t)MROWS * HID_KPAD), blk, 0, stream>>>(hpre, hg);

  // 9) FC2 + residual2 -> d_out
  gemm_bf16_wmma<EPI_RES><<<dim3(MROWS / BM, DMODEL / BN), blk, 0, stream>>>(
      hg, wfc2, fc2_b, x1, gamma2, out, DMODEL, HID_KPAD);
}
